// AttentionBlock_9363028705468
// MI455X (gfx1250) — matrix operand order repaired and checked by reading, not compiled, not run on hardware
//
#include <hip/hip_runtime.h>
#include <hip/hip_bf16.h>
#include <stdint.h>

typedef __attribute__((ext_vector_type(16))) _Float16 v16h;
typedef __attribute__((ext_vector_type(8)))  float    v8f;
typedef __attribute__((ext_vector_type(4)))  unsigned int v4u;
typedef __attribute__((ext_vector_type(8)))  unsigned int v8u;

#define NTHREADS 256

__device__ __forceinline__ v8f wmma_f16(v16h a, v16h b, v8f c) {
  // D = A(16x32 f16) * B(32x16 f16) + C(16x16 f32)
  return __builtin_amdgcn_wmma_f32_16x16x32_f16(
      /*neg_a=*/false, a, /*neg_b=*/false, b,
      /*c_mod=*/(short)0, c, /*reuse_a=*/false, /*reuse_b=*/false);
}

// A-fragment (16x32 f16). rowp points at 32 contiguous K halves for this lane's
// row (M = lane%16). Lane-half hi=0 holds K=[0..15], hi=1 holds K=[16..31]
// (same order as B fragments).
__device__ __forceinline__ v16h load_a_frag(const _Float16* rowp, int hi) {
  v16h a;
  const _Float16* p = rowp + 16 * hi;
#pragma unroll
  for (int j = 0; j < 16; ++j) a[j] = p[j];
  return a;
}

// B-fragment (32x16 f16) from an LDS tile stored K-contiguous per column:
// colp points at 32 contiguous K halves for this lane's column (N = lane%16).
// Lane-half hi=0 holds K=[0..15], hi=1 holds K=[16..31].
__device__ __forceinline__ v16h load_b_frag(const _Float16* colp, int hi) {
  v16h b;
  const _Float16* p = colp + 16 * hi;
#pragma unroll
  for (int j = 0; j < 16; ++j) b[j] = p[j];
  return b;
}

// ---------------------------------------------------------------------------
// GroupNorm(32, 512): block per (batch, group); 16 channels * 1024 spatial.
// ---------------------------------------------------------------------------
__global__ void gn_kernel(const float* __restrict__ x,
                          const float* __restrict__ gamma,
                          const float* __restrict__ beta,
                          _Float16* __restrict__ xn) {
  constexpr int CPG = 16, N = 1024, C = 512, G = 32;
  const int b = blockIdx.x / G;
  const int g = blockIdx.x % G;
  const float* xp = x + ((size_t)b * C + (size_t)g * CPG) * N;
  _Float16* op = xn + ((size_t)b * C + (size_t)g * CPG) * N;

  float s = 0.f, ss = 0.f;
  for (int i = threadIdx.x; i < CPG * N; i += NTHREADS) {
    float v = xp[i];
    s += v; ss += v * v;
  }
  __shared__ float red0[NTHREADS], red1[NTHREADS];
  red0[threadIdx.x] = s; red1[threadIdx.x] = ss;
  __syncthreads();
  for (int off = NTHREADS / 2; off > 0; off >>= 1) {
    if (threadIdx.x < off) {
      red0[threadIdx.x] += red0[threadIdx.x + off];
      red1[threadIdx.x] += red1[threadIdx.x + off];
    }
    __syncthreads();
  }
  const float inv_n = 1.0f / (CPG * N);
  const float mu = red0[0] * inv_n;
  const float var = red1[0] * inv_n - mu * mu;
  const float rs = rsqrtf(var + 1e-5f);

  for (int i = threadIdx.x; i < CPG * N; i += NTHREADS) {
    int c = g * CPG + i / N;
    float v = (xp[i] - mu) * rs * gamma[c] + beta[c];
    op[i] = (_Float16)v;
  }
}

// ---------------------------------------------------------------------------
// WMMA GEMM: out[b][m][n] = sum_k W[m][k] * X[b][k][n] + bias[m] (+residual)
// Block tile 64(M) x 128(N), K-step 64 (8 WMMAs per barrier round per wave).
// 8 waves as 2(M) x 4(N); each wave a 32x32 tile (2x2 wmma accumulators).
// ---------------------------------------------------------------------------
__global__ void gemm_wmma(const float* __restrict__ W,
                          const float* __restrict__ bias,
                          const _Float16* __restrict__ X,
                          _Float16* __restrict__ outH,
                          float* __restrict__ outF32,
                          const float* __restrict__ resid,
                          int M, int K, int N) {
  constexpr int BM = 64, BN = 128, BK = 64;
  __shared__ _Float16 Ws[BM][BK + 8];       // [m][k], 144B row stride (16B mult)
  __shared__ _Float16 Xs[BN][BK + 8];       // [n][k] (transposed for B frags)

  const int bb = blockIdx.z;
  const int m0 = blockIdx.y * BM;
  const int n0 = blockIdx.x * BN;
  const int t = threadIdx.x;
  const int wave = t / 32, lane = t % 32, l = lane % 16, hi = lane / 16;
  const int wm = wave / 4, wn = wave % 4;   // 2 x 4 wave grid

  const _Float16* Xb = X + (size_t)bb * K * N;
  v8f acc[2][2] = {};

  for (int k0 = 0; k0 < K; k0 += BK) {
    __syncthreads();
    {   // stage W tile 64x64 (f32 -> f16), 16 elems per thread
#pragma unroll
      for (int q = 0; q < 2; ++q) {
        int id = t + NTHREADS * q;          // 0..511
        int r = id / 8, c8 = (id % 8) * 8;
        const float* wp = W + (size_t)(m0 + r) * K + k0 + c8;
#pragma unroll
        for (int j = 0; j < 8; ++j) Ws[r][c8 + j] = (_Float16)wp[j];
      }
    }
    {   // stage X tile 64x128, transposed into Xs[n][k], 32 halves per thread
#pragma unroll
      for (int q = 0; q < 4; ++q) {
        int id = t + NTHREADS * q;          // 0..1023
        int kk = id / 16, c8 = (id % 16) * 8;
        const _Float16* xp = Xb + (size_t)(k0 + kk) * N + n0 + c8;
#pragma unroll
        for (int j = 0; j < 8; ++j) Xs[c8 + j][kk] = xp[j];
      }
    }
    if (k0 + BK < K)
      __builtin_prefetch(Xb + (size_t)(k0 + BK) * N + n0, 0, 1);
    __syncthreads();

#pragma unroll
    for (int kk = 0; kk < 2; ++kk) {
#pragma unroll
      for (int i = 0; i < 2; ++i) {
        v16h a = load_a_frag(&Ws[wm * 32 + i * 16 + l][kk * 32], hi);
#pragma unroll
        for (int jn = 0; jn < 2; ++jn) {
          v16h bf = load_b_frag(&Xs[wn * 32 + jn * 16 + l][kk * 32], hi);
          acc[i][jn] = wmma_f16(a, bf, acc[i][jn]);
        }
      }
    }
  }

  // epilogue: C layout -> (row = r + 8*hi, col = l) per 16x16 tile
#pragma unroll
  for (int i = 0; i < 2; ++i) {
#pragma unroll
    for (int jn = 0; jn < 2; ++jn) {
      int mbase = m0 + wm * 32 + i * 16 + 8 * hi;
      int ncol = n0 + wn * 32 + jn * 16 + l;
#pragma unroll
      for (int r = 0; r < 8; ++r) {
        int m = mbase + r;
        size_t idx = ((size_t)bb * M + m) * N + ncol;
        float v = acc[i][jn][r] + bias[m];
        if (outF32) outF32[idx] = resid[idx] + v;
        else        outH[idx] = (_Float16)v;
      }
    }
  }
}

// ---------------------------------------------------------------------------
// Flash attention: block per (b, h, 128 query rows). d=64, key tiles of 32.
// V tiles are fetched by the Tensor Data Mover (tensor_load_to_lds) with LDS
// row padding generated by the D# pad fields; K tiles are staged transposed
// by the VALU path. Each wave exclusively owns 16 query rows.
// ---------------------------------------------------------------------------
__global__ void attn_kernel(const _Float16* __restrict__ qkv,
                            _Float16* __restrict__ ha) {
  constexpr int C = 512, N = 1024, D = 64, NQ = 128, MT = 32;
  __shared__ _Float16 Qs[NQ][D + 8];        // [n][d], pre-scaled
  __shared__ _Float16 Ks[MT][D + 8];        // [m][d] (transposed)
  __shared__ _Float16 Vs[D][MT + 8];        // [d][m] (natural, TDM-filled)
  __shared__ _Float16 Ps[8][16][MT + 8];    // per-wave P tile [row][m]

  const int n0 = blockIdx.x * NQ;
  const int h = blockIdx.y, b = blockIdx.z;
  const int t = threadIdx.x, wave = t / 32, lane = t % 32;
  const int l = lane % 16, hi = lane / 16;
  const float scale = 0.125f;               // 64^-0.5

  const _Float16* q = qkv + ((size_t)b * 3 * C + (size_t)h * D) * N;
  const _Float16* kg = qkv + ((size_t)b * 3 * C + C + (size_t)h * D) * N;
  const _Float16* vg = qkv + ((size_t)b * 3 * C + 2 * C + (size_t)h * D) * N;

  // TDM descriptor group1 (constant per block): 2D tile 32(m) x 64(d) halves,
  // data_size=2B, pad_enable, pad_interval=16 DWORDs (code 3), pad=4 DWORDs
  // (code 3)  ->  LDS rows of 32 data halves + 8 pad halves == Vs[][40].
  const unsigned vs_lds = (unsigned)(uintptr_t)&Vs[0][0];
  v8u g1;
  g1.s0 = (1u << 16) | (1u << 20) | (3u << 22) | (3u << 25);
  g1.s1 = (unsigned)(N & 0xffff) << 16;               // tensor_dim0 = 1024
  g1.s2 = ((unsigned)N >> 16) | ((unsigned)D << 16);  // tensor_dim1 = 64
  g1.s3 = (unsigned)MT << 16;                         // tile_dim0 = 32
  g1.s4 = (unsigned)D;                                // tile_dim1 = 64
  g1.s5 = (unsigned)N;                                // tensor_dim0_stride
  g1.s6 = 0u;
  g1.s7 = 0u;

  // stage Q transposed + pre-scaled: Qs[n][d] = q[d][n0+n] * scale
  for (int id = t; id < D * NQ; id += NTHREADS) {
    int d = id / NQ, n = id % NQ;
    Qs[n][d] = (_Float16)((float)q[(size_t)d * N + n0 + n] * scale);
  }

  float mrow[8], lrow[8];
#pragma unroll
  for (int r = 0; r < 8; ++r) { mrow[r] = -3.0e38f; lrow[r] = 0.f; }
  v8f o[4] = {};                            // 16 rows x 64 d (4 col tiles)

  __syncthreads();

  for (int m0 = 0; m0 < N; m0 += MT) {
    if (wave == 0) {
      // TDM group0: count=1, lds_addr, 57-bit global_addr, type=2
      uint64_t ga = (uint64_t)(uintptr_t)(vg + m0);
      v4u g0;
      g0.x = 1u;
      g0.y = vs_lds;
      g0.z = (unsigned)(ga & 0xffffffffu);
      g0.w = ((unsigned)(ga >> 32) & 0x01ffffffu) | (2u << 30);
      asm volatile("tensor_load_to_lds %0, %1" :: "s"(g0), "s"(g1) : "memory");
    }
    {   // stage K tile transposed Ks[m][d] (8 halves per thread)
      int d = t / 4, c8 = (t % 4) * 8;
      const _Float16* kp = kg + (size_t)d * N + m0 + c8;
#pragma unroll
      for (int j = 0; j < 8; ++j) Ks[c8 + j][d] = kp[j];
    }
    if (wave == 0) __builtin_amdgcn_s_wait_tensorcnt(0);
    __syncthreads();

    // S tile: wave rows [wave*16 .. +15] x key cols [m0 .. m0+31]
    v8f s[2] = {};
#pragma unroll
    for (int kk = 0; kk < 2; ++kk) {        // K = d = 64 -> two 32-steps
      v16h a = load_a_frag(&Qs[wave * 16 + l][kk * 32], hi);
#pragma unroll
      for (int jn = 0; jn < 2; ++jn) {
        v16h bf = load_b_frag(&Ks[jn * 16 + l][kk * 32], hi);
        s[jn] = wmma_f16(a, bf, s[jn]);
      }
    }

    // online softmax; a lane holds rows (8*hi + r), cols (l + 16*jn)
    float p0[8], p1[8];
#pragma unroll
    for (int r = 0; r < 8; ++r) {
      float mx = fmaxf(s[0][r], s[1][r]);
#pragma unroll
      for (int off = 1; off < 16; off <<= 1)
        mx = fmaxf(mx, __shfl_xor(mx, off, 16));
      float mnew = fmaxf(mrow[r], mx);
      float alpha = __expf(mrow[r] - mnew);
      float e0 = __expf(s[0][r] - mnew);
      float e1 = __expf(s[1][r] - mnew);
      float rs = e0 + e1;
#pragma unroll
      for (int off = 1; off < 16; off <<= 1)
        rs += __shfl_xor(rs, off, 16);
      lrow[r] = lrow[r] * alpha + rs;
      mrow[r] = mnew;
      p0[r] = e0; p1[r] = e1;
#pragma unroll
      for (int td = 0; td < 4; ++td) o[td][r] *= alpha;
    }

    // re-layout P (C-layout) -> A-fragment layout via per-wave LDS tile
#pragma unroll
    for (int r = 0; r < 8; ++r) {
      Ps[wave][8 * hi + r][l]      = (_Float16)p0[r];
      Ps[wave][8 * hi + r][16 + l] = (_Float16)p1[r];
    }
    __syncthreads();

    // O += P(16x32) * V^T-layout(32x64): B frags from natural Vs[d][m]
    v16h pa = load_a_frag(&Ps[wave][l][0], hi);
#pragma unroll
    for (int td = 0; td < 4; ++td) {
      v16h bf = load_b_frag(&Vs[td * 16 + l][0], hi);
      o[td] = wmma_f16(pa, bf, o[td]);
    }
    __syncthreads();                        // before Ks/Vs are overwritten
  }

  // normalize and store ha[b][h*64 + d][n] as f16 for the proj GEMM
#pragma unroll
  for (int td = 0; td < 4; ++td) {
    int d = td * 16 + l;
#pragma unroll
    for (int r = 0; r < 8; ++r) {
      int n = n0 + wave * 16 + 8 * hi + r;
      float val = o[td][r] / lrow[r];
      ha[((size_t)b * C + (size_t)h * D + d) * N + n] = (_Float16)val;
    }
  }
}

// ---------------------------------------------------------------------------
extern "C" void kernel_launch(void* const* d_in, const int* in_sizes, int n_in,
                              void* d_out, int out_size, void* d_ws, size_t ws_size,
                              hipStream_t stream) {
  (void)in_sizes; (void)n_in; (void)out_size; (void)ws_size;
  constexpr int B = 8, C = 512, N = 1024;

  const float* x        = (const float*)d_in[0];
  const float* gn_gamma = (const float*)d_in[1];
  const float* gn_beta  = (const float*)d_in[2];
  const float* w_qkv    = (const float*)d_in[3];
  const float* b_qkv    = (const float*)d_in[4];
  const float* w_proj   = (const float*)d_in[5];
  const float* b_proj   = (const float*)d_in[6];
  float* out = (float*)d_out;

  // workspace layout (f16): xn [B*C*N], qkv [B*3C*N], ha [B*C*N]
  char* ws = (char*)d_ws;
  _Float16* xn  = (_Float16*)ws;                                   // 8 MB
  _Float16* qkv = (_Float16*)(ws + (size_t)B * C * N * 2);         // 24 MB
  _Float16* hav = (_Float16*)(ws + (size_t)B * C * N * 2 * 4);     // 8 MB

  // 1) GroupNorm -> f16
  gn_kernel<<<dim3(B * 32), dim3(NTHREADS), 0, stream>>>(x, gn_gamma, gn_beta, xn);

  // 2) QKV GEMM: M=1536, K=512, N=1024 (f16 out, +bias)
  gemm_wmma<<<dim3(N / 128, (3 * C) / 64, B), dim3(NTHREADS), 0, stream>>>(
      w_qkv, b_qkv, xn, qkv, nullptr, nullptr, 3 * C, C, N);

  // 3) Flash attention: per (b, h, 128-row query chunk)
  attn_kernel<<<dim3(N / 128, 8, B), dim3(NTHREADS), 0, stream>>>(qkv, hav);

  // 4) proj GEMM + bias + residual (fp32 out)
  gemm_wmma<<<dim3(N / 128, C / 64, B), dim3(NTHREADS), 0, stream>>>(
      w_proj, b_proj, hav, nullptr, out, x, C, C, N);
}
